// EntropyWeightNetwork_13872744366386
// MI455X (gfx1250) — matrix operand order repaired and checked by reading, not compiled, not run on hardware
//
#include <hip/hip_runtime.h>
#include <hip/hip_bf16.h>

typedef __attribute__((ext_vector_type(16))) __bf16 v16bf;
typedef __attribute__((ext_vector_type(8)))  float  v8f;

__device__ __forceinline__ unsigned short f2bf(float x) {
    __bf16 b = (__bf16)x;                   // native v_cvt RNE on gfx1250
    return __builtin_bit_cast(unsigned short, b);
}
__device__ __forceinline__ float bf2f(unsigned short h) {
    return __uint_as_float(((unsigned)h) << 16);
}

#define BM 128
#define BN 64
#define BK 32

// ---------------------------------------------------------------------------
// Kernel 1: fold k_embed and pos_enc (row-constant features) + b1 into a
// single effective bias vector of length 256.
// ---------------------------------------------------------------------------
__global__ __launch_bounds__(256) void setup_bias_kernel(
    const float* __restrict__ W1, const float* __restrict__ b1,
    const int* __restrict__ pk, const int* __restrict__ pt,
    const int* __restrict__ pf, const int* __restrict__ ps,
    const int* __restrict__ pms, float* __restrict__ ebias)
{
    int j = threadIdx.x;                    // 0..255 output column
    float acc = b1[j];
    float kk = (float)pk[0];
    const float neg_log = -9.21034037197618f / 31.0f;  // -ln(10000)/(half-1)
    #pragma unroll 8
    for (int i = 0; i < 64; ++i) {
        int h = (i < 32) ? i : (i - 32);
        float e = kk * __expf((float)h * neg_log);
        float val = (i < 32) ? __sinf(e) : __cosf(e);
        acc += val * W1[(size_t)(4096 + i) * 256 + j];
    }
    float tf = (float)pt[0], ff = (float)pf[0];
    float pe0 = __sinf(0.1f * tf), pe1 = __cosf(0.1f * tf);
    float pe2 = __sinf(0.1f * ff), pe3 = __cosf(0.1f * ff);
    float pe4 = (float)ps[0] / (float)pms[0];
    acc += pe0 * W1[(size_t)4160 * 256 + j];
    acc += pe1 * W1[(size_t)4161 * 256 + j];
    acc += pe2 * W1[(size_t)4162 * 256 + j];
    acc += pe3 * W1[(size_t)4163 * 256 + j];
    acc += pe4 * W1[(size_t)4164 * 256 + j];
    ebias[j] = acc;
}

// ---------------------------------------------------------------------------
// Kernel 2: per-row statistics. One 256-thread workgroup per row of 4096.
// Median (lower median, index 2047) via 4-pass radix select over monotone
// uint keys kept resident in LDS.
// ---------------------------------------------------------------------------
__global__ __launch_bounds__(256) void stats_kernel(
    const float* __restrict__ Z, float* __restrict__ stats)
{
    __shared__ unsigned keys[4096];
    __shared__ unsigned hist[256];
    __shared__ float rs[256], rq[256], r1[256], rmn[256], rmx[256], rp[256], rn[256];
    __shared__ unsigned s_prefix;
    __shared__ int s_k;

    int row = blockIdx.x, t = threadIdx.x;
    const float* z = Z + (size_t)row * 4096;

    float sum = 0.f, sq = 0.f, l1 = 0.f, pos = 0.f, neg = 0.f;
    float mn = 3.402823e38f, mx = -3.402823e38f;
    #pragma unroll 4
    for (int i = 0; i < 16; ++i) {
        float v = z[t + i * 256];
        sum += v; sq += v * v; l1 += fabsf(v);
        mn = fminf(mn, v); mx = fmaxf(mx, v);
        pos += (v > 0.f) ? 1.f : 0.f;
        neg += (v < 0.f) ? 1.f : 0.f;
        unsigned u = __float_as_uint(v);
        keys[t + i * 256] = (u & 0x80000000u) ? ~u : (u | 0x80000000u);
    }
    rs[t] = sum; rq[t] = sq; r1[t] = l1; rmn[t] = mn; rmx[t] = mx; rp[t] = pos; rn[t] = neg;
    __syncthreads();
    for (int st = 128; st > 0; st >>= 1) {
        if (t < st) {
            rs[t] += rs[t + st]; rq[t] += rq[t + st]; r1[t] += r1[t + st];
            rmn[t] = fminf(rmn[t], rmn[t + st]);
            rmx[t] = fmaxf(rmx[t], rmx[t + st]);
            rp[t] += rp[t + st]; rn[t] += rn[t + st];
        }
        __syncthreads();
    }

    if (t == 0) { s_prefix = 0u; s_k = 2047; }   // (4096-1)//2 lower median
    __syncthreads();

    for (int p = 3; p >= 0; --p) {
        hist[t] = 0u;
        __syncthreads();
        unsigned hmask = (p == 3) ? 0u : (0xFFFFFFFFu << ((p + 1) * 8));
        unsigned pref = s_prefix;
        for (int i = 0; i < 16; ++i) {
            unsigned kk = keys[t + i * 256];
            if ((kk & hmask) == (pref & hmask))
                atomicAdd(&hist[(kk >> (p * 8)) & 255u], 1u);
        }
        __syncthreads();
        if (t == 0) {
            int kk = s_k;
            unsigned b = 0;
            for (b = 0; b < 256; ++b) {
                int h = (int)hist[b];
                if (kk < h) break;
                kk -= h;
            }
            s_prefix |= (b << (p * 8));
            s_k = kk;
        }
        __syncthreads();
    }

    if (t == 0) {
        float mean = rs[0] * (1.f / 4096.f);
        float var = (rq[0] - 4096.f * mean * mean) * (1.f / 4095.f);
        var = fmaxf(var, 0.f);
        unsigned key = s_prefix;
        unsigned u = (key & 0x80000000u) ? (key ^ 0x80000000u) : ~key;
        float med = __uint_as_float(u);
        float* o = stats + (size_t)row * 10;
        o[0] = mean; o[1] = sqrtf(var); o[2] = rmn[0]; o[3] = rmx[0]; o[4] = med;
        o[5] = var;  o[6] = sqrtf(rq[0]); o[7] = r1[0]; o[8] = rp[0]; o[9] = rn[0];
    }
}

// ---------------------------------------------------------------------------
// Kernel 3: layer-1 GEMM. A = z_local f32 (converted to bf16 into LDS),
// B = W1[0:4096,0:256] f32 (converted into LDS). 128x64 tile per block,
// 8 waves x (16 rows x 64 cols), v_wmma_f32_16x16x32_bf16.
// global_prefetch of the next A K-tile overlaps HBM latency with WMMA.
// Epilogue adds effective bias + stats@W1[4165:4175] and applies relu.
// ---------------------------------------------------------------------------
__global__ __launch_bounds__(256) void gemm1_kernel(
    const float* __restrict__ Z, const float* __restrict__ stats,
    const float* __restrict__ W1, const float* __restrict__ ebias,
    unsigned short* __restrict__ H1)
{
    __shared__ unsigned short sA[BM * BK];   // [m][k] row-major
    __shared__ unsigned short sB[BN * BK];   // [n][k]
    const int K = 4096, N = 256;
    int mBase = blockIdx.x * BM;
    int nBase = blockIdx.y * BN;
    int tid = threadIdx.x;
    int wave = tid >> 5, lane = tid & 31;
    int half = lane >> 4, l16 = lane & 15;

    v8f acc[4] = {};

    for (int k0 = 0; k0 < K; k0 += BK) {
        // prefetch next K-tile of A (one cacheline probe per thread)
        if (k0 + BK < K) {
            const float* nf = Z + (size_t)(mBase + (tid >> 1)) * K + (k0 + BK) + (tid & 1) * 16;
            __builtin_prefetch(nf, 0, 3);
        }
        // stage A tile: 128x32 f32 -> bf16 (1024 float4 chunks, 4 per thread)
        #pragma unroll
        for (int i = 0; i < 4; ++i) {
            int c = tid + i * 256;
            int arow = c >> 3, cir = c & 7;
            const float4 v = *(const float4*)(Z + (size_t)(mBase + arow) * K + k0 + cir * 4);
            uint2 pkd;
            pkd.x = ((unsigned)f2bf(v.x)) | (((unsigned)f2bf(v.y)) << 16);
            pkd.y = ((unsigned)f2bf(v.z)) | (((unsigned)f2bf(v.w)) << 16);
            *(uint2*)&sA[arow * BK + cir * 4] = pkd;
        }
        // stage B tile: 32x64 slice of W1, stored [n][k]
        #pragma unroll
        for (int i = 0; i < 8; ++i) {
            int e = tid + i * 256;
            int n = e & 63, kk = e >> 6;
            sB[n * BK + kk] = f2bf(W1[(size_t)(k0 + kk) * N + nBase + n]);
        }
        __syncthreads();

        union FragA { v16bf v; uint4 q[2]; } fa;
        const uint4* pA = (const uint4*)&sA[(wave * 16 + l16) * BK];
        fa.q[0] = pA[half];        // K = half*8 .. +7
        fa.q[1] = pA[2 + half];    // K = 16+half*8 .. +7
        #pragma unroll
        for (int j = 0; j < 4; ++j) {
            union FragB { v16bf v; uint4 q[2]; } fbv;
            const uint4* pB = (const uint4*)&sB[(j * 16 + l16) * BK];
            fbv.q[0] = pB[half]; fbv.q[1] = pB[2 + half];
            acc[j] = __builtin_amdgcn_wmma_f32_16x16x32_bf16(
                false, fa.v, false, fbv.v, (short)0, acc[j], false, false);
        }
        __syncthreads();
    }

    float ebj[4];
    #pragma unroll
    for (int j = 0; j < 4; ++j) ebj[j] = ebias[nBase + j * 16 + l16];
    const float* W1s = W1 + (size_t)4165 * N;   // stats rows of W1
    #pragma unroll
    for (int r = 0; r < 8; ++r) {
        int m = mBase + wave * 16 + half * 8 + r;
        float st[10];
        #pragma unroll
        for (int i = 0; i < 10; ++i) st[i] = stats[(size_t)m * 10 + i];
        #pragma unroll
        for (int j = 0; j < 4; ++j) {
            int n = nBase + j * 16 + l16;
            float v = acc[j][r] + ebj[j];
            #pragma unroll
            for (int i = 0; i < 10; ++i)
                v += st[i] * W1s[(size_t)i * N + n];
            v = v > 0.f ? v : 0.f;
            H1[(size_t)m * N + n] = f2bf(v);
        }
    }
}

// ---------------------------------------------------------------------------
// Kernel 4: middle layers GEMM (A already bf16). out = relu(A@W + b), bf16.
// A tile is a straight bf16 copy -> use gfx1250 async global->LDS DMA
// (global_load_async_to_lds_b128, ASYNCcnt-tracked, bypasses VGPRs).
// ---------------------------------------------------------------------------
__global__ __launch_bounds__(256) void gemm_mid_kernel(
    const unsigned short* __restrict__ A, int K,
    const float* __restrict__ Wt, int N,
    const float* __restrict__ bias,
    unsigned short* __restrict__ C)
{
    __shared__ unsigned short sA[BM * BK];
    __shared__ unsigned short sB[BN * BK];
    int mBase = blockIdx.x * BM;
    int nBase = blockIdx.y * BN;
    int tid = threadIdx.x;
    int wave = tid >> 5, lane = tid & 31;
    int half = lane >> 4, l16 = lane & 15;

    v8f acc[4] = {};

    for (int k0 = 0; k0 < K; k0 += BK) {
        // stage A: 128x32 bf16 = 512 b128 chunks, 2 per thread, async DMA to LDS
        #pragma unroll
        for (int i = 0; i < 2; ++i) {
            int c = tid + i * 256;
            int arow = c >> 2, cir = c & 3;
            unsigned ldsA = (unsigned)(size_t)&sA[arow * BK + cir * 8];
            unsigned long long ga =
                (unsigned long long)(A + (size_t)(mBase + arow) * K + k0 + cir * 8);
            asm volatile("global_load_async_to_lds_b128 %0, %1, off"
                         :: "v"(ldsA), "v"(ga) : "memory");
        }
        // stage B: 32xBN slice of W (f32 row-major [K][N]) -> [n][k] bf16
        #pragma unroll
        for (int i = 0; i < 8; ++i) {
            int e = tid + i * 256;
            int n = e & 63, kk = e >> 6;
            sB[n * BK + kk] = f2bf(Wt[(size_t)(k0 + kk) * N + nBase + n]);
        }
        asm volatile("s_wait_asynccnt 0x0" ::: "memory");
        __syncthreads();

        union FragA { v16bf v; uint4 q[2]; } fa;
        const uint4* pA = (const uint4*)&sA[(wave * 16 + l16) * BK];
        fa.q[0] = pA[half];
        fa.q[1] = pA[2 + half];
        #pragma unroll
        for (int j = 0; j < 4; ++j) {
            union FragB { v16bf v; uint4 q[2]; } fbv;
            const uint4* pB = (const uint4*)&sB[(j * 16 + l16) * BK];
            fbv.q[0] = pB[half]; fbv.q[1] = pB[2 + half];
            acc[j] = __builtin_amdgcn_wmma_f32_16x16x32_bf16(
                false, fa.v, false, fbv.v, (short)0, acc[j], false, false);
        }
        __syncthreads();
    }

    #pragma unroll
    for (int j = 0; j < 4; ++j) {
        int n = nBase + j * 16 + l16;
        float b = bias[n];
        #pragma unroll
        for (int r = 0; r < 8; ++r) {
            int m = mBase + wave * 16 + half * 8 + r;
            float v = acc[j][r] + b;
            v = v > 0.f ? v : 0.f;
            C[(size_t)m * N + n] = f2bf(v);
        }
    }
}

// ---------------------------------------------------------------------------
// Kernel 5: head 64->5 mat-vec + fused softmax, one thread per row.
// ---------------------------------------------------------------------------
__global__ __launch_bounds__(256) void head_kernel(
    const unsigned short* __restrict__ H3,
    const float* __restrict__ W4, const float* __restrict__ b4,
    float* __restrict__ out, int M)
{
    int r = blockIdx.x * 256 + threadIdx.x;
    if (r >= M) return;
    float lg[5];
    #pragma unroll
    for (int j = 0; j < 5; ++j) lg[j] = b4[j];
    const unsigned short* h = H3 + (size_t)r * 64;
    #pragma unroll 8
    for (int i = 0; i < 64; ++i) {
        float hv = bf2f(h[i]);
        #pragma unroll
        for (int j = 0; j < 5; ++j) lg[j] += hv * W4[i * 5 + j];
    }
    float mx = lg[0];
    #pragma unroll
    for (int j = 1; j < 5; ++j) mx = fmaxf(mx, lg[j]);
    float s = 0.f;
    #pragma unroll
    for (int j = 0; j < 5; ++j) { lg[j] = __expf(lg[j] - mx); s += lg[j]; }
    float inv = 1.f / s;
    #pragma unroll
    for (int j = 0; j < 5; ++j) out[(size_t)r * 5 + j] = lg[j] * inv;
}

// ---------------------------------------------------------------------------
extern "C" void kernel_launch(void* const* d_in, const int* in_sizes, int n_in,
                              void* d_out, int out_size, void* d_ws, size_t ws_size,
                              hipStream_t stream)
{
    const float* Z   = (const float*)d_in[0];
    const int*   pk  = (const int*)d_in[1];
    const int*   pt  = (const int*)d_in[2];
    const int*   pf  = (const int*)d_in[3];
    const int*   ps  = (const int*)d_in[4];
    const int*   pms = (const int*)d_in[5];
    const float* W1  = (const float*)d_in[6];
    const float* b1  = (const float*)d_in[7];
    const float* W2  = (const float*)d_in[8];
    const float* b2  = (const float*)d_in[9];
    const float* W3  = (const float*)d_in[10];
    const float* b3  = (const float*)d_in[11];
    const float* W4  = (const float*)d_in[12];
    const float* b4  = (const float*)d_in[13];

    int M = in_sizes[0] / 4096;   // 8192 rows (B * prod(DATA_SHAPE) / 4096)

    // workspace layout
    float* stats = (float*)d_ws;                         // M*10 f32
    float* ebias = stats + (size_t)M * 10;               // 256 f32
    unsigned short* h1 = (unsigned short*)(ebias + 256); // M*256 bf16
    unsigned short* h2 = h1 + (size_t)M * 256;           // M*128 bf16
    unsigned short* h3 = h2 + (size_t)M * 128;           // M*64  bf16

    setup_bias_kernel<<<1, 256, 0, stream>>>(W1, b1, pk, pt, pf, ps, pms, ebias);
    stats_kernel<<<M, 256, 0, stream>>>(Z, stats);
    gemm1_kernel<<<dim3(M / BM, 256 / BN), 256, 0, stream>>>(Z, stats, W1, ebias, h1);
    gemm_mid_kernel<<<dim3(M / BM, 128 / BN), 256, 0, stream>>>(h1, 256, W2, 128, b2, h2);
    gemm_mid_kernel<<<dim3(M / BM, 64 / BN), 256, 0, stream>>>(h2, 128, W3, 64, b3, h3);
    head_kernel<<<(M + 255) / 256, 256, 0, stream>>>(h3, W4, b4, (float*)d_out, M);
}
